// SWAAttention_46737834115402
// MI455X (gfx1250) — matrix operand order repaired and checked by reading, not compiled, not run on hardware
//
#include <hip/hip_runtime.h>

typedef __attribute__((ext_vector_type(16))) __bf16 v16bf;
typedef __attribute__((ext_vector_type(8)))  __bf16 v8bf;
typedef __attribute__((ext_vector_type(8)))  float  v8f;

#define B_ 2
#define T_ 2048
#define C_ 2048
#define H_ 16
#define W_ 512
#define D_ 128
#define NEGINF (-1.0e30f)

// Build a 16-element bf16 WMMA fragment from two contiguous 8-element pieces.
__device__ inline v16bf frag16(const __bf16* p0, const __bf16* p1) {
  v8bf lo = *(const v8bf*)p0;
  v8bf hi = *(const v8bf*)p1;
  return __builtin_shufflevector(lo, hi, 0,1,2,3,4,5,6,7,8,9,10,11,12,13,14,15);
}

__device__ inline v8f wmma_bf16(v16bf a, v16bf b, v8f c) {
  // (neg_a, A, neg_b, B, c_mod, C, reuse_a, reuse_b)
  return __builtin_amdgcn_wmma_f32_16x16x32_bf16(false, a, false, b, (short)0, c, false, false);
}

// CDNA5 async global->LDS copy (ASYNCcnt-tracked). VDST lane value = LDS byte
// address; generic pointers to __shared__ keep the LDS offset in bits [31:0].
__device__ inline unsigned lds_off(const void* p) {
  return (unsigned)(unsigned long long)p;
}
__device__ inline void async_ld16(unsigned dst_lds, const void* gsrc) {
  asm volatile("global_load_async_to_lds_b128 %0, %1, off"
               :: "v"(dst_lds), "v"(gsrc) : "memory");
}
__device__ inline void wait_async() {
  asm volatile("s_wait_asynccnt 0x0" ::: "memory");
}

// ---------------- elementwise fp32 -> bf16 ----------------
__global__ void k_cvt(const float* __restrict__ src, __bf16* __restrict__ dst, int n) {
  int i = blockIdx.x * blockDim.x + threadIdx.x;
  if (i < n) dst[i] = (__bf16)src[i];
}

// ---------------- tiled transpose + convert: dst[n][k] = (bf16)src[k][n] ----------------
__global__ __launch_bounds__(256)
void k_transpose_cvt(const float* __restrict__ src, __bf16* __restrict__ dst,
                     int rows, int cols) {
  __shared__ float tile[32][33];
  int c0 = blockIdx.x * 32, r0 = blockIdx.y * 32;
  #pragma unroll
  for (int j = 0; j < 4; ++j) {
    int r = threadIdx.y + j * 8;
    tile[r][threadIdx.x] = src[(size_t)(r0 + r) * cols + c0 + threadIdx.x];
  }
  __syncthreads();
  #pragma unroll
  for (int j = 0; j < 4; ++j) {
    int n = threadIdx.y + j * 8;
    dst[(size_t)(c0 + n) * rows + r0 + threadIdx.x] = (__bf16)tile[threadIdx.x][n];
  }
}

// ---------------- bf16 GEMM: C[M,N] = A[M,K] * Bt[N,K]^T ----------------
// 256 threads = 8 waves in a 4(M) x 2(N) grid; block tile 128x128; each wave
// computes 32x64 (8 WMMA per 32-K stage). K staged 32 at a time, double
// buffered in LDS, filled with async global->LDS loads overlapped with WMMA.
template <bool OUT_BF16>
__global__ __launch_bounds__(256)
void k_gemm(const __bf16* __restrict__ A, const __bf16* __restrict__ Bt,
            void* __restrict__ Cout, int M, int N, int K) {
  __shared__ alignas(16) __bf16 As[2][128 * 32];  // 8 KB per buffer
  __shared__ alignas(16) __bf16 Bs[2][128 * 32];  // 8 KB per buffer
  const int tid   = threadIdx.x;
  const int wave  = tid >> 5;
  const int lane  = tid & 31;
  const int half  = (lane >> 4) & 1;
  const int lm    = lane & 15;
  const int waveM = wave >> 1;   // 0..3
  const int waveN = wave & 1;    // 0..1
  const int n0 = blockIdx.x * 128;
  const int m0 = blockIdx.y * 128;

  v8f acc[2][4];
  const v8f z = {0,0,0,0,0,0,0,0};
  #pragma unroll
  for (int s = 0; s < 2; ++s)
    #pragma unroll
    for (int ns = 0; ns < 4; ++ns) acc[s][ns] = z;

  // Each stage moves A tile 128x32 + B tile 128x32 = 1024 x 16B chunks,
  // 4 async chunk loads per thread (2 A + 2 B).
  auto issue = [&](int buf, int kc) {
    #pragma unroll
    for (int j = 0; j < 2; ++j) {
      int c = tid + j * 256;              // 0..511
      int row = c >> 2, off = (c & 3) * 8;
      async_ld16(lds_off(&As[buf][row * 32 + off]),
                 &A[(size_t)(m0 + row) * K + kc + off]);
      async_ld16(lds_off(&Bs[buf][row * 32 + off]),
                 &Bt[(size_t)(n0 + row) * K + kc + off]);
    }
  };

  issue(0, 0);
  int buf = 0;
  for (int kc = 0; kc < K; kc += 32) {
    wait_async();        // this wave's fills of `buf` have landed in LDS
    __syncthreads();     // everyone's fills landed; previous compute finished
    if (kc + 32 < K) issue(buf ^ 1, kc + 32);

    // A fragments (16x32 layout): lane<16 -> K {0-7,16-23}; lane>=16 -> {8-15,24-31}
    int ar0 = (waveM * 32 + lm) * 32 + half * 8;
    int ar1 = (waveM * 32 + 16 + lm) * 32 + half * 8;
    v16bf af0 = frag16(&As[buf][ar0], &As[buf][ar0 + 16]);
    v16bf af1 = frag16(&As[buf][ar1], &As[buf][ar1 + 16]);
    #pragma unroll
    for (int ns = 0; ns < 4; ++ns) {
      // B fragment (32x16): lane = N column; lane<16 -> K {0-7,16-23}; lane>=16 -> {8-15,24-31}
      int brow = (waveN * 64 + ns * 16 + lm) * 32 + half * 8;
      v16bf bf = frag16(&Bs[buf][brow], &Bs[buf][brow + 16]);
      acc[0][ns] = wmma_bf16(af0, bf, acc[0][ns]);
      acc[1][ns] = wmma_bf16(af1, bf, acc[1][ns]);
    }
    buf ^= 1;
  }

  // C/D layout: VGPR r -> row r (lanes 0-15) / row r+8 (lanes 16-31), lane%16 = col
  #pragma unroll
  for (int s = 0; s < 2; ++s)
    #pragma unroll
    for (int ns = 0; ns < 4; ++ns)
      #pragma unroll
      for (int r = 0; r < 8; ++r) {
        int gm = m0 + waveM * 32 + s * 16 + r + half * 8;
        int gn = n0 + waveN * 64 + ns * 16 + lm;
        float v = acc[s][ns][r];
        if (OUT_BF16) ((__bf16*)Cout)[(size_t)gm * N + gn] = (__bf16)v;
        else          ((float*)Cout)[(size_t)gm * N + gn] = v;
      }
}

// ---------------- RoPE + head split; V stored transposed [b,h,d,t] ----------------
__global__ void k_rope(const __bf16* __restrict__ qkv,
                       const float* __restrict__ fc, const float* __restrict__ fs,
                       __bf16* __restrict__ Qr, __bf16* __restrict__ Kr,
                       __bf16* __restrict__ Vt) {
  int i = blockIdx.x * blockDim.x + threadIdx.x;   // B*H*T*(D/2) = 2^22 threads
  int d = i & 63;
  int t = (i >> 6) & (T_ - 1);
  int h = (i >> 17) & (H_ - 1);
  int b = i >> 21;
  if (b >= B_) return;
  size_t base = ((size_t)b * T_ + t) * (3 * C_) + (size_t)h * D_;
  float q1 = (float)qkv[base + d],            q2 = (float)qkv[base + d + 64];
  float k1 = (float)qkv[base + C_ + d],       k2 = (float)qkv[base + C_ + d + 64];
  float v1 = (float)qkv[base + 2 * C_ + d],   v2 = (float)qkv[base + 2 * C_ + d + 64];
  float c1 = fc[t * D_ + d],      s1 = fs[t * D_ + d];
  float c2 = fc[t * D_ + d + 64], s2 = fs[t * D_ + d + 64];
  const float sc = 0.08838834764831845f;  // 1/sqrt(128), folded into Q
  float qa = (q1 * c1 - q2 * s1) * sc;
  float qb = (q2 * c2 + q1 * s2) * sc;
  float ka = k1 * c1 - k2 * s1;
  float kb = k2 * c2 + k1 * s2;
  size_t ob = ((size_t)(b * H_ + h) * T_ + t) * D_;
  Qr[ob + d] = (__bf16)qa; Qr[ob + d + 64] = (__bf16)qb;
  Kr[ob + d] = (__bf16)ka; Kr[ob + d + 64] = (__bf16)kb;
  size_t vb = (size_t)(b * H_ + h) * D_ * (size_t)T_;
  Vt[vb + (size_t)d * T_ + t]        = (__bf16)v1;
  Vt[vb + (size_t)(d + 64) * T_ + t] = (__bf16)v2;
}

// ---------------- banded flash attention: one wave per 16-query tile ----------------
__global__ __launch_bounds__(32)
void k_attn(const __bf16* __restrict__ Qr, const __bf16* __restrict__ Kr,
            const __bf16* __restrict__ Vt, __bf16* __restrict__ Y) {
  __shared__ alignas(16) __bf16 Pl[16 * 32];  // P staging for layout transpose
  const int lane = threadIdx.x;
  const int half = lane >> 4;
  const int lm   = lane & 15;
  const int q0 = blockIdx.x * 16;
  const int bh = blockIdx.y;
  const int b = bh >> 4, h = bh & 15;

  const __bf16* Qbase = Qr + (size_t)bh * T_ * D_;
  const __bf16* Kbase = Kr + (size_t)bh * T_ * D_;
  const __bf16* Vbase = Vt + (size_t)bh * D_ * (size_t)T_;

  // Q fragments (A layout), loaded once, reused across the whole key loop.
  v16bf qf[4];
  {
    const __bf16* qrow = Qbase + (size_t)(q0 + lm) * D_;
    #pragma unroll
    for (int dc = 0; dc < 4; ++dc) {
      int o = dc * 32 + half * 8;
      qf[dc] = frag16(qrow + o, qrow + o + 16);
    }
  }

  float mrow[8], lrow[8];
  v8f acc[8];
  const v8f z = {0,0,0,0,0,0,0,0};
  #pragma unroll
  for (int r = 0; r < 8; ++r) { mrow[r] = NEGINF; lrow[r] = 0.f; }
  #pragma unroll
  for (int nt = 0; nt < 8; ++nt) acc[nt] = z;

  int ks = q0 - W_ + 1; if (ks < 0) ks = 0; ks &= ~31;

  for (int kb = ks; kb < q0 + 16; kb += 32) {
    // ---- S = Q * K^T for 32 keys (two 16-key halves), K=128 -> 4 chained WMMA each
    v8f s01[2];
    #pragma unroll
    for (int hh = 0; hh < 2; ++hh) {
      int k0 = kb + hh * 16;
      const __bf16* krow = Kbase + (size_t)(k0 + lm) * D_;
      v8f s = z;
      #pragma unroll
      for (int dc = 0; dc < 4; ++dc) {
        int o = dc * 32 + half * 8;
        v16bf kf = frag16(krow + o, krow + o + 16);
        s = wmma_bf16(qf[dc], kf, s);
      }
      // band mask: valid iff (j <= i) && (j > i - W)
      int j = k0 + lm;
      #pragma unroll
      for (int r = 0; r < 8; ++r) {
        int iq = q0 + r + half * 8;
        if (j > iq || j + W_ <= iq) s[r] = NEGINF;
      }
      s01[hh] = s;
    }
    // ---- online softmax: per-row stats via 16-lane butterfly (rows live in halves)
    float scale[8];
    #pragma unroll
    for (int r = 0; r < 8; ++r) {
      float t = fmaxf(s01[0][r], s01[1][r]);
      #pragma unroll
      for (int m = 1; m < 16; m <<= 1) t = fmaxf(t, __shfl_xor(t, m, 32));
      float nm = fmaxf(mrow[r], t);
      scale[r] = __expf(mrow[r] - nm);
      mrow[r] = nm;
      float p0 = __expf(s01[0][r] - nm);
      float p1 = __expf(s01[1][r] - nm);
      float rs = p0 + p1;
      #pragma unroll
      for (int m = 1; m < 16; m <<= 1) rs += __shfl_xor(rs, m, 32);
      lrow[r] = lrow[r] * scale[r] + rs;
      int row = r + half * 8;                    // C-layout -> LDS row-major [16][32]
      Pl[row * 32 + lm]      = (__bf16)p0;
      Pl[row * 32 + 16 + lm] = (__bf16)p1;
    }
    #pragma unroll
    for (int nt = 0; nt < 8; ++nt)
      #pragma unroll
      for (int r = 0; r < 8; ++r) acc[nt][r] *= scale[r];

    // same-wave LDS RAW: wait for the b16 stores before fragment reads
    asm volatile("s_wait_dscnt 0" ::: "memory");

    // P fragment in A layout
    v16bf pf = frag16(&Pl[lm * 32 + half * 8], &Pl[lm * 32 + half * 8 + 16]);

    // ---- O += P * V : 8 n-subtiles cover D=128; Vt rows contiguous along t
    #pragma unroll
    for (int nt = 0; nt < 8; ++nt) {
      const __bf16* vrow = Vbase + (size_t)(nt * 16 + lm) * T_ + kb + half * 8;
      v16bf vf = frag16(vrow, vrow + 16);
      acc[nt] = wmma_bf16(pf, vf, acc[nt]);
    }
  }

  // normalize and scatter to Y[b,t,h*D+d] (bf16 input for the projection GEMM)
  float rl[8];
  #pragma unroll
  for (int r = 0; r < 8; ++r) rl[r] = 1.0f / lrow[r];
  #pragma unroll
  for (int nt = 0; nt < 8; ++nt)
    #pragma unroll
    for (int r = 0; r < 8; ++r) {
      int t = q0 + r + half * 8;
      int d = nt * 16 + lm;
      Y[((size_t)b * T_ + t) * C_ + h * D_ + d] = (__bf16)(acc[nt][r] * rl[r]);
    }
}

extern "C" void kernel_launch(void* const* d_in, const int* in_sizes, int n_in,
                              void* d_out, int out_size, void* d_ws, size_t ws_size,
                              hipStream_t stream) {
  (void)in_sizes; (void)n_in; (void)out_size; (void)ws_size;
  const float* x      = (const float*)d_in[0];
  const float* fc     = (const float*)d_in[1];
  const float* fs     = (const float*)d_in[2];
  const float* w_attn = (const float*)d_in[3];
  const float* w_proj = (const float*)d_in[4];

  char* ws = (char*)d_ws;
  size_t o = 0;
  auto alloc = [&](size_t bytes) {
    char* p = ws + o;
    o += (bytes + 255) & ~(size_t)255;
    return p;
  };
  __bf16* xb  = (__bf16*)alloc((size_t)B_ * T_ * C_ * 2);
  __bf16* wat = (__bf16*)alloc((size_t)3 * C_ * C_ * 2);      // w_attn^T [3C][C]
  __bf16* wpt = (__bf16*)alloc((size_t)C_ * C_ * 2);          // w_proj^T [C][C]
  __bf16* qkv = (__bf16*)alloc((size_t)B_ * T_ * 3 * C_ * 2);
  __bf16* Qr  = (__bf16*)alloc((size_t)B_ * H_ * T_ * D_ * 2);
  __bf16* Kr  = (__bf16*)alloc((size_t)B_ * H_ * T_ * D_ * 2);
  __bf16* Vt  = (__bf16*)alloc((size_t)B_ * H_ * T_ * D_ * 2 + 256);
  __bf16* Y   = (__bf16*)alloc((size_t)B_ * T_ * C_ * 2);

  int nx = B_ * T_ * C_;
  k_cvt<<<(nx + 255) / 256, 256, 0, stream>>>(x, xb, nx);
  k_transpose_cvt<<<dim3(3 * C_ / 32, C_ / 32), dim3(32, 8), 0, stream>>>(w_attn, wat, C_, 3 * C_);
  k_transpose_cvt<<<dim3(C_ / 32, C_ / 32), dim3(32, 8), 0, stream>>>(w_proj, wpt, C_, C_);

  k_gemm<true><<<dim3(3 * C_ / 128, (B_ * T_) / 128), 256, 0, stream>>>(
      xb, wat, qkv, B_ * T_, 3 * C_, C_);

  int nr = B_ * H_ * T_ * (D_ / 2);
  k_rope<<<nr / 256, 256, 0, stream>>>(qkv, fc, fs, Qr, Kr, Vt);

  k_attn<<<dim3(T_ / 16, B_ * H_), 32, 0, stream>>>(Qr, Kr, Vt, Y);

  k_gemm<false><<<dim3(C_ / 128, (B_ * T_) / 128), 256, 0, stream>>>(
      Y, wpt, d_out, B_ * T_, C_, C_);
}
